// SetConvEncoder_88880053223887
// MI455X (gfx1250) — matrix operand order repaired and checked by reading, not compiled, not run on hardware
//
#include <hip/hip_runtime.h>

typedef __attribute__((ext_vector_type(16))) _Float16 v16h;
typedef __attribute__((ext_vector_type(8)))  _Float16 v8h;
typedef __attribute__((ext_vector_type(8)))  float    v8f;
typedef __attribute__((ext_vector_type(8)))  int      v8i;

// Problem constants (from reference setup_inputs)
constexpr int BATCH = 8;
constexpr int NCTX  = 2048;
constexpr int GH    = 128;
constexpr int GW    = 128;
constexpr int HW    = GH * GW;
constexpr int COUT  = 4;                 // 3 signal channels + 1 density

constexpr int KC    = 256;               // context points staged in LDS per chunk
constexpr int WAVES = 8;                 // waves per block (wave32 -> 256 threads)
constexpr int TPB   = WAVES * 32;
constexpr int GPTS_PER_BLOCK = WAVES * 16;          // 128 grid points per block
constexpr int BLOCKS_PER_B   = HW / GPTS_PER_BLOCK; // 128

constexpr int YSTRIDE = KC + 8;          // halves; 528B rows: 16B-aligned + bank skew

__global__ __launch_bounds__(TPB)
void setconv_wmma_kernel(const float* __restrict__ xc,    // [B,N,2]
                         const float* __restrict__ yc,    // [B,N,3]
                         const float* __restrict__ mask,  // [B,N]
                         const float* __restrict__ logls, // [1]
                         const float* __restrict__ grid,  // [H,W,2] (x,y)
                         float* __restrict__ out)         // [B,4,H,W]
{
    // Staged, pre-scaled context data:
    //   s_sx0[k] = -2c*x0[k], s_sx1[k] = -2c*x1[k], s_c0[k] = c*(x0^2+x1^2)
    // so exp2 arg = fma(gx,sx0, fma(gy,sx1, c0)) + c*|g|^2      (c = -0.5*log2e/ls^2)
    __shared__ __align__(16) float    s_sx0[KC];
    __shared__ __align__(16) float    s_sx1[KC];
    __shared__ __align__(16) float    s_c0 [KC];
    // Transposed B stage: rows 0..2 = y*mask, row 3 = mask, row 4 = zeros (cols 4..15)
    __shared__ __align__(16) _Float16 s_yT[(COUT + 1) * YSTRIDE];

    const int tid  = threadIdx.x;
    const int wave = tid >> 5;
    const int lane = tid & 31;
    const int h    = lane >> 4;          // half-group (K sub-range selector)
    const int m    = lane & 15;          // A row within tile / B,D column

    const int b        = blockIdx.x / BLOCKS_PER_B;
    const int blockInB = blockIdx.x % BLOCKS_PER_B;
    const int gbase    = blockInB * GPTS_PER_BLOCK + wave * 16;

    // This lane's grid point (A-tile row); both half-groups share it.
    const int   g  = gbase + m;
    const float gx = grid[g * 2 + 0];
    const float gy = grid[g * 2 + 1];

    // c = -0.5*log2(e)/ls^2, computed divide-free: 1/ls^2 = exp2(-2*logls*log2e)
    const float L2E  = 1.44269504088896340736f;
    const float cexp = -0.5f * L2E * __builtin_amdgcn_exp2f(-2.0f * L2E * logls[0]);
    const float bsl  = cexp * (gx * gx + gy * gy);   // per-lane constant term
    const float m2c  = -2.0f * cexp;

    const float* xb = xc   + (size_t)b * NCTX * 2;
    const float* yb = yc   + (size_t)b * NCTX * 3;
    const float* mb = mask + (size_t)b * NCTX;

    // Zero-fill the dummy B row. Only entries 0..KC-1 are ever read
    // (k0+8h+23 <= 255), so one unconditional store per thread suffices.
    s_yT[COUT * YSTRIDE + tid] = (_Float16)0.f;

    const int yrow = (m < COUT) ? m : COUT;          // lanes m>=4 read the zero row

    v8f acc = {};  // 16x16 f32 accumulator (cols 0..3 meaningful)

    for (int kc0 = 0; kc0 < NCTX; kc0 += KC) {
        __syncthreads();
        // Stage KC context points: one per thread (TPB == KC).
        {
            const int k  = tid;
            const int kg = kc0 + k;
            const float x0 = xb[kg * 2 + 0];
            const float x1 = xb[kg * 2 + 1];
            s_sx0[k] = m2c * x0;
            s_sx1[k] = m2c * x1;
            s_c0 [k] = cexp * __builtin_fmaf(x0, x0, x1 * x1);
            const float mv = mb[kg];
            s_yT[0 * YSTRIDE + k] = (_Float16)(yb[kg * 3 + 0] * mv);
            s_yT[1 * YSTRIDE + k] = (_Float16)(yb[kg * 3 + 1] * mv);
            s_yT[2 * YSTRIDE + k] = (_Float16)(yb[kg * 3 + 2] * mv);
            s_yT[3 * YSTRIDE + k] = (_Float16)mv;
        }
        __syncthreads();

#pragma unroll
        for (int k0 = 0; k0 < KC; k0 += 32) {
            // ---- Build A (16x32 f16 weights), ISA A-layout ----
            // lane(h,m): run 0 covers K = k0+8h..+7      -> packed dwords 0..3
            //            run 1 covers K = k0+16+8h..+7   -> packed dwords 4..7
            // All 16 lanes of a half-group read identical LDS addresses (broadcast).
            v8i apack;
#pragma unroll
            for (int run = 0; run < 2; ++run) {
                const int kb = k0 + 16 * run + 8 * h;   // 8 consecutive K, 32B-aligned
                const float4 X0a = *(const float4*)&s_sx0[kb];
                const float4 X0b = *(const float4*)&s_sx0[kb + 4];
                const float4 X1a = *(const float4*)&s_sx1[kb];
                const float4 X1b = *(const float4*)&s_sx1[kb + 4];
                const float4 C0a = *(const float4*)&s_c0[kb];
                const float4 C0b = *(const float4*)&s_c0[kb + 4];
                const float x0[8] = {X0a.x, X0a.y, X0a.z, X0a.w, X0b.x, X0b.y, X0b.z, X0b.w};
                const float x1[8] = {X1a.x, X1a.y, X1a.z, X1a.w, X1b.x, X1b.y, X1b.z, X1b.w};
                const float c0[8] = {C0a.x, C0a.y, C0a.z, C0a.w, C0b.x, C0b.y, C0b.z, C0b.w};
#pragma unroll
                for (int jj = 0; jj < 4; ++jj) {
                    const float w0 = __builtin_amdgcn_exp2f(
                        __builtin_fmaf(gx, x0[2 * jj + 0],
                        __builtin_fmaf(gy, x1[2 * jj + 0], c0[2 * jj + 0])) + bsl);
                    const float w1 = __builtin_amdgcn_exp2f(
                        __builtin_fmaf(gx, x0[2 * jj + 1],
                        __builtin_fmaf(gy, x1[2 * jj + 1], c0[2 * jj + 1])) + bsl);
                    // v_cvt_pk_f16_f32: both halves packed in one instruction
                    apack[4 * run + jj] =
                        __builtin_bit_cast(int, __builtin_amdgcn_cvt_pkrtz(w0, w1));
                }
            }
            const v16h a = __builtin_bit_cast(v16h, apack);

            // ---- Build B (32x16 f16): lane=col n=m holds K = k0+8h+j (j<8), k0+16+8h+(j-8) (j>=8)
            // Two 16-byte runs per lane -> two ds_load_b128 (broadcast per col).
            const _Float16* src = &s_yT[yrow * YSTRIDE + k0 + 8 * h];
            const v8h blo = *(const v8h*)(src);
            const v8h bhi = *(const v8h*)(src + 16);
            const v16h bm = __builtin_shufflevector(blo, bhi,
                0, 1, 2, 3, 4, 5, 6, 7, 8, 9, 10, 11, 12, 13, 14, 15);

            // EXEC is all-1s here (no divergence in this loop).
            acc = __builtin_amdgcn_wmma_f32_16x16x32_f16(
                /*neg_a=*/false, a, /*neg_b=*/false, bm,
                /*c_mod=*/(short)0, acc, /*reuse_a=*/false, /*reuse_b=*/false);
        }
    }

    // ---- Store D: lane l, VGPR r -> row (r + 8h), col m. Cols 0..3 only. ----
    // Rows r..r+3 are consecutive grid points -> two global_store_b128 per lane.
    if (m < COUT) {
        float* dst = out + ((size_t)b * COUT + m) * HW + gbase + 8 * h;
        const float4 o0 = {acc[0], acc[1], acc[2], acc[3]};
        const float4 o1 = {acc[4], acc[5], acc[6], acc[7]};
        *(float4*)(dst)     = o0;
        *(float4*)(dst + 4) = o1;
    }
}

extern "C" void kernel_launch(void* const* d_in, const int* in_sizes, int n_in,
                              void* d_out, int out_size, void* d_ws, size_t ws_size,
                              hipStream_t stream) {
    const float* xc    = (const float*)d_in[0];  // [8,2048,2]
    const float* yc    = (const float*)d_in[1];  // [8,2048,3]
    const float* msk   = (const float*)d_in[2];  // [8,2048]
    const float* logls = (const float*)d_in[3];  // [1]
    const float* grid  = (const float*)d_in[4];  // [128,128,2]
    float*       out   = (float*)d_out;          // [8,4,128,128]

    const int nblocks = BATCH * BLOCKS_PER_B;    // 1024
    setconv_wmma_kernel<<<nblocks, TPB, 0, stream>>>(xc, yc, msk, logls, grid, out);
}
